// LSTM_25220047962689
// MI455X (gfx1250) — matrix operand order repaired and checked by reading, not compiled, not run on hardware
//
#include <hip/hip_runtime.h>

// ============================================================================
// LSTM forward on MI455X (gfx1250), wave32 + WMMA.
//
// Key algorithmic facts exploited:
//  * VOCAB == 3  =>  the 550-GFLOP input projection xe@Wx collapses to a
//    [3 x 2048] table lookup: table[v] = emb[v] @ [W_gx|W_ix|W_fx|W_ox] + b.
//  * Batch rows are independent in the recurrence => partition batch across
//    16 self-contained workgroups (16 rows each); h/c state never leaves the
//    WGP, so the only sync is one __syncthreads() per time step.
//  * Wh (2 MB as bf16, stored transposed) is L2-resident on a 192 MB L2 and
//    streamed per step; per-wave WMMA work: 2 subtiles x 4 gates x 16 k-steps
//    = 128 x v_wmma_f32_16x16x32_bf16 per step.
//
// Fragment layouts used here (A and B share contiguous k order; C/D per CDNA5 ISA 7.12.2):
//  * A (16x32 bf16, M=batch,K=h): lane l -> row m=l&15, K chunks
//    {half*16 .. half*16+7} and {half*16+8 .. +15}  => two 16B ds_read_b128.
//  * B (32x16 bf16, K=h,N=gatecol): lane l -> col n=l&15, K run
//    {half*16 .. half*16+15} => 32 contiguous bytes of WhT[col][k]
//    => two global_load_b128.
//  * C/D (16x16 f32): VGPR v, lane l -> (m = v + 8*(l>>4), n = l&15).
// ============================================================================

#define SEQ_LEN     512
#define EMB_DIM     1024
#define HIDDEN      512
#define GCOLS       2048     // 4 * HIDDEN
#define NUM_CLASSES 10
#define BATCH       256

#define NWG         16       // workgroups (batch tiles)
#define BT          16       // batch rows per workgroup
#define NTHREADS    512      // 16 waves of 32
#define HPAD        520      // h row stride in bf16 elems (bank-conflict pad)

typedef __attribute__((ext_vector_type(16))) __bf16 v16bf;
typedef __attribute__((ext_vector_type(8)))  __bf16 v8bf;
typedef __attribute__((ext_vector_type(8)))  float  v8f;

union V16 { v16bf v; v8bf h[2]; };

__device__ __forceinline__ float sigm_f(float x) { return 1.0f / (1.0f + __expf(-x)); }
__device__ __forceinline__ float tanh_f(float x) { return 2.0f / (1.0f + __expf(-2.0f * x)) - 1.0f; }

// ---------------------------------------------------------------------------
// P1: table[v][g*512+j] = b_g[j] + sum_e emb[v][e] * W_{g}x[e][j]   (f32)
// 3*2048 outputs, 1024-deep dot each (~13 MFLOP total).
// ---------------------------------------------------------------------------
__global__ void build_table(const float* __restrict__ emb,
                            const float* __restrict__ Wgx, const float* __restrict__ Wix,
                            const float* __restrict__ Wfx, const float* __restrict__ Wox,
                            const float* __restrict__ bg,  const float* __restrict__ bi,
                            const float* __restrict__ bf,  const float* __restrict__ bo,
                            float* __restrict__ table) {
    int idx = blockIdx.x * blockDim.x + threadIdx.x;
    if (idx >= 3 * GCOLS) return;
    int v = idx / GCOLS, col = idx % GCOLS;
    int g = col >> 9, j = col & 511;
    const float* W = (g == 0) ? Wgx : (g == 1) ? Wix : (g == 2) ? Wfx : Wox;
    const float* B = (g == 0) ? bg  : (g == 1) ? bi  : (g == 2) ? bf  : bo;
    const float* e = emb + v * EMB_DIM;
    float acc = B[j];
    for (int k = 0; k < EMB_DIM; ++k) acc += e[k] * W[k * HIDDEN + j];
    table[idx] = acc;
}

// ---------------------------------------------------------------------------
// P2: WhT[g*512 + j][k] = (bf16) W_{g}h[k][j]   — transposed bf16 pack so that
// WMMA B-fragments are contiguous 32-byte runs (global_load_b128 x2 per frag).
// ---------------------------------------------------------------------------
__global__ void pack_whT(const float* __restrict__ Wgh, const float* __restrict__ Wih,
                         const float* __restrict__ Wfh, const float* __restrict__ Woh,
                         __bf16* __restrict__ WhT) {
    int idx = blockIdx.x * blockDim.x + threadIdx.x;
    if (idx >= GCOLS * HIDDEN) return;
    int r = idx >> 9, k = idx & 511;       // r = g*512 + j
    int g = r >> 9, j = r & 511;
    const float* W = (g == 0) ? Wgh : (g == 1) ? Wih : (g == 2) ? Wfh : Woh;
    WhT[idx] = (__bf16)W[k * HIDDEN + j];
}

// ---------------------------------------------------------------------------
// Main persistent LSTM kernel: 16 WGs x 512 threads (16 waves).
// Wave w owns h columns [32w, 32w+32) as two 16-col subtiles; per subtile it
// accumulates all four gate tiles (g,i,f,o) via WMMA, applies the cell update
// in f32 registers, and writes its bf16 h slice into the ping-pong LDS buffer.
// ---------------------------------------------------------------------------
__global__ __launch_bounds__(NTHREADS) void
lstm_persistent(const int* __restrict__ x, const float* __restrict__ table,
                const __bf16* __restrict__ WhT,
                const float* __restrict__ Wph, const float* __restrict__ bp,
                float* __restrict__ out) {
    __shared__ __align__(16) __bf16 hbuf[2][BT][HPAD];   // 33280 B
    __shared__ float tableL[3 * GCOLS];                  // 24576 B
    __shared__ int   tokL[2][BT];                        //   128 B

    const int tid  = threadIdx.x;
    const int lane = tid & 31;
    const int wv   = tid >> 5;           // wave 0..15
    const int m16  = lane & 15;          // A row / B col / C col index
    const int half = lane >> 4;          // lane half-select
    const int b0   = blockIdx.x * BT;    // first batch row of this WG

    // One-time staging: zero h ping-pong, copy gate-input table, first tokens.
    for (int i = tid; i < 2 * BT * HPAD; i += NTHREADS)
        (&hbuf[0][0][0])[i] = (__bf16)0.0f;
    for (int i = tid; i < 3 * GCOLS; i += NTHREADS)
        tableL[i] = table[i];
    if (tid < BT) tokL[0][tid] = x[(b0 + tid) * SEQ_LEN];
    __syncthreads();

    // Persistent f32 cell state: 2 subtiles x 8 accumulator regs per lane.
    float c0[8], c1[8];
#pragma unroll
    for (int v = 0; v < 8; ++v) { c0[v] = 0.0f; c1[v] = 0.0f; }

#pragma unroll 1
    for (int t = 0; t < SEQ_LEN; ++t) {
        const int cur = t & 1;
        const int nxt = cur ^ 1;
        const __bf16* Arow = &hbuf[cur][m16][half * 16];

#pragma unroll
        for (int s = 0; s < 2; ++s) {
            const int jb = wv * 32 + s * 16;         // gate-column subtile base
            float (&cst)[8] = s ? c1 : c0;

            // B stream base pointers (row fixed per gate; advance 32 elems/kt)
            const __bf16* pG = WhT + (size_t)(0 * HIDDEN + jb + m16) * HIDDEN + half * 16;
            const __bf16* pI = WhT + (size_t)(1 * HIDDEN + jb + m16) * HIDDEN + half * 16;
            const __bf16* pF = WhT + (size_t)(2 * HIDDEN + jb + m16) * HIDDEN + half * 16;
            const __bf16* pO = WhT + (size_t)(3 * HIDDEN + jb + m16) * HIDDEN + half * 16;

            // C init = xg_t (token-gathered input projection incl. bias)
            v8f accG, accI, accF, accO;
#pragma unroll
            for (int v = 0; v < 8; ++v) {
                const int m  = v + half * 8;
                const int tk = tokL[cur][m];
                const int bA = tk * GCOLS + jb + m16;
                accG[v] = tableL[bA];
                accI[v] = tableL[bA + 512];
                accF[v] = tableL[bA + 1024];
                accO[v] = tableL[bA + 1536];
            }

            // K reduction: 16 x (1 A-frag from LDS, 4 B-frags from L2, 4 WMMA)
#pragma unroll 4
            for (int kt = 0; kt < 16; ++kt) {
                V16 A, Bg, Bi, Bf, Bo;
                const __bf16* ap = Arow + kt * 32;
                A.h[0] = *(const v8bf*)(ap);
                A.h[1] = *(const v8bf*)(ap + 8);
                const int ko = kt * 32;
                Bg.h[0] = *(const v8bf*)(pG + ko); Bg.h[1] = *(const v8bf*)(pG + ko + 8);
                Bi.h[0] = *(const v8bf*)(pI + ko); Bi.h[1] = *(const v8bf*)(pI + ko + 8);
                Bf.h[0] = *(const v8bf*)(pF + ko); Bf.h[1] = *(const v8bf*)(pF + ko + 8);
                Bo.h[0] = *(const v8bf*)(pO + ko); Bo.h[1] = *(const v8bf*)(pO + ko + 8);

                accG = __builtin_amdgcn_wmma_f32_16x16x32_bf16(false, A.v, false, Bg.v,
                                                               (short)0, accG, false, false);
                accI = __builtin_amdgcn_wmma_f32_16x16x32_bf16(false, A.v, false, Bi.v,
                                                               (short)0, accI, false, false);
                accF = __builtin_amdgcn_wmma_f32_16x16x32_bf16(false, A.v, false, Bf.v,
                                                               (short)0, accF, false, false);
                accO = __builtin_amdgcn_wmma_f32_16x16x32_bf16(false, A.v, false, Bo.v,
                                                               (short)0, accO, false, false);
            }

            // Cell update in f32 registers; write bf16 h slice to ping buffer.
#pragma unroll
            for (int v = 0; v < 8; ++v) {
                const int m = v + half * 8;
                float g = tanh_f(accG[v]);
                float i = sigm_f(accI[v]);
                float f = sigm_f(accF[v]);
                float o = sigm_f(accO[v]);
                float c = g * i + cst[v] * f;
                cst[v] = c;
                hbuf[nxt][m][jb + m16] = (__bf16)(tanh_f(c) * o);
            }
        }

        // Stage next step's tokens into the same ping-pong phase as h.
        if (tid < BT) {
            int tn = (t + 1 < SEQ_LEN) ? (t + 1) : (SEQ_LEN - 1);
            tokL[nxt][tid] = x[(b0 + tid) * SEQ_LEN + tn];
        }
        __syncthreads();   // single barrier per step (ping-pong buffers)
    }

    // Final h lives in hbuf[0] (512 toggles). Projection + log_softmax,
    // one thread per batch row (tiny: 16 x 512 x 10 MACs per WG).
    if (tid < BT) {
        const int m = tid;
        float logits[NUM_CLASSES];
#pragma unroll
        for (int c = 0; c < NUM_CLASSES; ++c) logits[c] = bp[c];
        for (int k = 0; k < HIDDEN; ++k) {
            float hk = (float)hbuf[0][m][k];
#pragma unroll
            for (int c = 0; c < NUM_CLASSES; ++c)
                logits[c] += hk * Wph[k * NUM_CLASSES + c];
        }
        float mx = logits[0];
#pragma unroll
        for (int c = 1; c < NUM_CLASSES; ++c) mx = fmaxf(mx, logits[c]);
        float se = 0.0f;
#pragma unroll
        for (int c = 0; c < NUM_CLASSES; ++c) se += __expf(logits[c] - mx);
        float lse = mx + __logf(se);
#pragma unroll
        for (int c = 0; c < NUM_CLASSES; ++c)
            out[(b0 + m) * NUM_CLASSES + c] = logits[c] - lse;
    }
}

// ---------------------------------------------------------------------------
// Host entry. Workspace layout: [0,24KB) f32 table, [32KB, 32KB+2MB) bf16 WhT.
// ---------------------------------------------------------------------------
extern "C" void kernel_launch(void* const* d_in, const int* in_sizes, int n_in,
                              void* d_out, int out_size, void* d_ws, size_t ws_size,
                              hipStream_t stream) {
    const int*   x   = (const int*)d_in[0];
    const float* emb = (const float*)d_in[1];
    const float* Wgx = (const float*)d_in[2];
    const float* Wgh = (const float*)d_in[3];
    const float* bg  = (const float*)d_in[4];
    const float* Wix = (const float*)d_in[5];
    const float* Wih = (const float*)d_in[6];
    const float* bi  = (const float*)d_in[7];
    const float* Wfx = (const float*)d_in[8];
    const float* Wfh = (const float*)d_in[9];
    const float* bf  = (const float*)d_in[10];
    const float* Wox = (const float*)d_in[11];
    const float* Woh = (const float*)d_in[12];
    const float* bo  = (const float*)d_in[13];
    const float* Wph = (const float*)d_in[14];
    const float* bp  = (const float*)d_in[15];
    float* out = (float*)d_out;

    float*  table = (float*)d_ws;
    __bf16* WhT   = (__bf16*)((char*)d_ws + 32768);

    build_table<<<(3 * GCOLS + 255) / 256, 256, 0, stream>>>(
        emb, Wgx, Wix, Wfx, Wox, bg, bi, bf, bo, table);
    pack_whT<<<(GCOLS * HIDDEN) / 256, 256, 0, stream>>>(
        Wgh, Wih, Wfh, Woh, WhT);
    lstm_persistent<<<NWG, NTHREADS, 0, stream>>>(
        x, table, WhT, Wph, bp, out);
}
